// AdaptiveMultiGabor2DLayer_36481452212321
// MI455X (gfx1250) — matrix operand order repaired and checked by reading, not compiled, not run on hardware
//
#include <hip/hip_runtime.h>

// ---------------------------------------------------------------------------
// AdaptiveMultiGabor2D for MI455X (gfx1250, wave32, WMMA).
//
// Reconstruct per-channel complex weights once (cheap, 3.4e7 cmuls) into bf16
// re/im planes in d_ws, then a complex GEMM [2048x256]x[256x256] per
// channel/set with v_wmma_f32_16x16x32_bf16 and a fused exp/sincos epilogue.
// ~137 GFLOP matrix work, ~536 MB HBM traffic (~23 us at 23.3 TB/s), weights
// (33.5 MB bf16) resident in the 192 MB L2. B tiles are staged with
// GLOBAL_LOAD_ASYNC_TO_LDS_B128 (ASYNCcnt path); A tiles go through VGPRs for
// the f32->bf16 conversion.
// ---------------------------------------------------------------------------

typedef __bf16 bf16_t;
typedef bf16_t v8bf  __attribute__((ext_vector_type(8)));
typedef bf16_t v16bf __attribute__((ext_vector_type(16)));
typedef float  v8f   __attribute__((ext_vector_type(8)));
typedef unsigned int v8u __attribute__((ext_vector_type(8)));
typedef int    v4i   __attribute__((ext_vector_type(4)));

typedef __attribute__((address_space(1))) v4i g_v4i;   // global int4
typedef __attribute__((address_space(3))) v4i l_v4i;   // LDS int4

#define IN_F  256
#define OUT_F 256
#define NCHAN 64
#define RANK  4
#define NPTS  2048
#define WPLANE ((size_t)NCHAN * IN_F * OUT_F)   // elems per weight plane

__device__ __forceinline__ v16bf cat16(v8bf lo, v8bf hi) {
    return __builtin_shufflevector(lo, hi, 0,1,2,3,4,5,6,7,8,9,10,11,12,13,14,15);
}

__device__ __forceinline__ v8f wmma_bf16(v16bf a, v16bf b, v8f c) {
    // D = A(16x32 bf16) * B(32x16 bf16) + C(16x16 f32)
    return __builtin_amdgcn_wmma_f32_16x16x32_bf16(
        false, a, false, b, (short)0, c, false, false);
}

// ---------------------------------------------------------------------------
// Kernel 1: Wall[m][c] = sum_r U_set[idx[c],r] * V_set[r]  (complex),
// written TRANSPOSED as bf16 planes Wall[m][c][n][k] (m: W1r,W1i,W2r,W2i) so
// the GEMM kernel's B fragments are contiguous-in-K. LDS transpose keeps both
// the V reads (contiguous in n) and W writes (contiguous in k) coalesced.
// grid.x = 64 channels * 16 (64x64 tiles of the 256x256 weight).
// ---------------------------------------------------------------------------
__global__ void __launch_bounds__(256)
gabor_build_weights(const int* __restrict__ idx,
                    const float2* __restrict__ U1, const float2* __restrict__ V1,
                    const float2* __restrict__ U2, const float2* __restrict__ V2,
                    bf16_t* __restrict__ Wall)
{
    __shared__ __align__(16) bf16_t t4[4][64 * 72];   // [plane][nn][kk], padded

    const int c    = blockIdx.x >> 4;
    const int tile = blockIdx.x & 15;
    const int k0   = (tile & 3) << 6;
    const int n0   = (tile >> 2) << 6;
    const int tid  = threadIdx.x;
    const int ci   = idx[c];

    float2 u1[RANK], u2[RANK];
#pragma unroll
    for (int r = 0; r < RANK; ++r) {
        u1[r] = U1[ci * RANK + r];
        u2[r] = U2[ci * RANK + r];
    }

#pragma unroll
    for (int i = 0; i < 16; ++i) {
        int lin = tid + (i << 8);          // 0..4095
        int kk  = lin >> 6;                // 0..63
        int nn  = lin & 63;                // fast index -> coalesced V reads
        size_t off = (size_t)(k0 + kk) * OUT_F + (n0 + nn);   // V[r, in*256+out]
        float ar = 0.f, ai = 0.f, br = 0.f, bi = 0.f;
#pragma unroll
        for (int r = 0; r < RANK; ++r) {
            float2 v = V1[(size_t)r * (IN_F * OUT_F) + off];
            ar += u1[r].x * v.x - u1[r].y * v.y;
            ai += u1[r].x * v.y + u1[r].y * v.x;
            float2 w = V2[(size_t)r * (IN_F * OUT_F) + off];
            br += u2[r].x * w.x - u2[r].y * w.y;
            bi += u2[r].x * w.y + u2[r].y * w.x;
        }
        int la = nn * 72 + kk;             // transpose in LDS
        t4[0][la] = (bf16_t)ar;
        t4[1][la] = (bf16_t)ai;
        t4[2][la] = (bf16_t)br;
        t4[3][la] = (bf16_t)bi;
    }
    __syncthreads();

    struct BP { bf16_t lo, hi; };
#pragma unroll
    for (int m = 0; m < 4; ++m) {
#pragma unroll
        for (int j = 0; j < 8; ++j) {
            int lin = tid + (j << 8);      // 0..2047 uint-pairs
            int nn  = lin >> 5;            // 0..63
            int k   = (lin & 31) << 1;     // 0..62 even -> k contiguous writes
            BP bp { t4[m][nn * 72 + k], t4[m][nn * 72 + k + 1] };
            unsigned int pk = __builtin_bit_cast(unsigned int, bp);
            *(unsigned int*)&Wall[(size_t)m * WPLANE + (size_t)c * (IN_F * OUT_F)
                                  + (size_t)(n0 + nn) * IN_F + k0 + k] = pk;
        }
    }
}

// ---------------------------------------------------------------------------
// Kernel 2: complex GEMM + fused Gabor epilogue.
// Block tile M=128 x N=32, K-step 32. 8 waves; wave w owns rows [16w,16w+16),
// two 16x16 N-subtiles, 4 f32 accumulators each (Yr1,Yi1,Yr2,Yi2).
// Per K-step/subtile: Yr = Xr@Wr + (-Xi)@Wi ; Yi = Xr@Wi + Xi@Wr  (8 WMMAs).
// B tiles staged via async-to-LDS (ASYNCcnt); A tiles via VGPR + cvt.
// grid.x = 64 channels * 16 M-tiles, grid.y = 8 N-tiles.
// ---------------------------------------------------------------------------
__global__ void __launch_bounds__(256, 2)
gabor_main(const float2* __restrict__ x,  const int* __restrict__ idx,
           const float2* __restrict__ b1, const float2* __restrict__ b2,
           const bf16_t* __restrict__ Wall,
           float2* __restrict__ out)
{
    const int LDA = 40;                    // 32 + 8 pad (bank-conflict free)
    const int LDB = 40;
    __shared__ __align__(16) bf16_t Asr[128 * 40];
    __shared__ __align__(16) bf16_t Asi[128 * 40];
    __shared__ __align__(16) bf16_t Bs[4][32 * 40];

    const int c     = blockIdx.x >> 4;
    const int mBase = (blockIdx.x & 15) << 7;
    const int nBase = blockIdx.y << 5;
    const int tid   = threadIdx.x;
    const int wave  = tid >> 5;
    const int lane  = tid & 31;
    const int half  = lane >> 4;           // 0: lanes 0-15, 1: lanes 16-31
    const int lr16  = lane & 15;

    const size_t wchan = (size_t)c * (IN_F * OUT_F);

    v8f acc[2][4];
#pragma unroll
    for (int i = 0; i < 2; ++i)
#pragma unroll
        for (int j = 0; j < 4; ++j)
            acc[i][j] = v8f{0.f, 0.f, 0.f, 0.f, 0.f, 0.f, 0.f, 0.f};

    for (int kt = 0; kt < 8; ++kt) {
        const int kBase = kt << 5;

        // ---- stage B tiles: 4 planes of [32n x 32k] bf16, async to LDS ----
        // 512 x 16B transfers; thread t2 -> plane m = t2>>7, row n, 16B chunk.
#pragma unroll
        for (int j = 0; j < 2; ++j) {
            int t2  = tid + (j << 8);          // 0..511
            int m   = t2 >> 7;                 // plane
            int rem = t2 & 127;
            int n   = rem >> 2;
            int k   = (rem & 3) << 3;          // elem offset, 16B aligned
            const bf16_t* g = Wall + (size_t)m * WPLANE + wchan
                            + (size_t)(nBase + n) * IN_F + kBase + k;
            __builtin_amdgcn_global_load_async_to_lds_b128(
                (g_v4i*)g, (l_v4i*)&Bs[m][n * LDB + k], 0, 0);
        }

        // ---- stage A tile: x[c, mBase..+128, kBase..+32] -> bf16 re/im ----
#pragma unroll
        for (int i = 0; i < 16; ++i) {
            int lin = tid + (i << 8);          // 0..4095
            int row = lin >> 5;
            int k   = lin & 31;                // fast -> coalesced 8B loads
            float2 v = x[((size_t)c * NPTS + mBase + row) * IN_F + kBase + k];
            Asr[row * LDA + k] = (bf16_t)v.x;
            Asi[row * LDA + k] = (bf16_t)v.y;
        }

        __builtin_amdgcn_s_wait_asynccnt(0);   // async LDS writes visible
        __syncthreads();

        // ---- A fragments, contiguous k order (k = 16*half + s), same as B ----
        const int arow = wave * 16 + lr16;
        v8bf alo, ahi;
        alo = *(const v8bf*)&Asr[arow * LDA + half * 16];
        ahi = *(const v8bf*)&Asr[arow * LDA + half * 16 + 8];
        v16bf xr = cat16(alo, ahi);
        alo = *(const v8bf*)&Asi[arow * LDA + half * 16];
        ahi = *(const v8bf*)&Asi[arow * LDA + half * 16 + 8];
        v16bf xi = cat16(alo, ahi);
        // -Xi via packed sign flip (bf16 NEG bits must be 0 in WMMA)
        v8u xb = __builtin_bit_cast(v8u, xi);
        xb = xb ^ 0x80008000u;
        v16bf xin = __builtin_bit_cast(v16bf, xb);

#pragma unroll
        for (int nt = 0; nt < 2; ++nt) {
            const int bn = nt * 16 + lr16;     // B column owned by this lane
            v16bf w0, w1, w2, w3;
            {
                v8bf lo = *(const v8bf*)&Bs[0][bn * LDB + half * 16];
                v8bf hi = *(const v8bf*)&Bs[0][bn * LDB + half * 16 + 8];
                w0 = cat16(lo, hi);
            }
            {
                v8bf lo = *(const v8bf*)&Bs[1][bn * LDB + half * 16];
                v8bf hi = *(const v8bf*)&Bs[1][bn * LDB + half * 16 + 8];
                w1 = cat16(lo, hi);
            }
            {
                v8bf lo = *(const v8bf*)&Bs[2][bn * LDB + half * 16];
                v8bf hi = *(const v8bf*)&Bs[2][bn * LDB + half * 16 + 8];
                w2 = cat16(lo, hi);
            }
            {
                v8bf lo = *(const v8bf*)&Bs[3][bn * LDB + half * 16];
                v8bf hi = *(const v8bf*)&Bs[3][bn * LDB + half * 16 + 8];
                w3 = cat16(lo, hi);
            }
            acc[nt][0] = wmma_bf16(xr,  w0, acc[nt][0]);   // Yr1 += Xr*W1r
            acc[nt][0] = wmma_bf16(xin, w1, acc[nt][0]);   // Yr1 -= Xi*W1i
            acc[nt][1] = wmma_bf16(xr,  w1, acc[nt][1]);   // Yi1 += Xr*W1i
            acc[nt][1] = wmma_bf16(xi,  w0, acc[nt][1]);   // Yi1 += Xi*W1r
            acc[nt][2] = wmma_bf16(xr,  w2, acc[nt][2]);   // Yr2
            acc[nt][2] = wmma_bf16(xin, w3, acc[nt][2]);
            acc[nt][3] = wmma_bf16(xr,  w3, acc[nt][3]);   // Yi2
            acc[nt][3] = wmma_bf16(xi,  w2, acc[nt][3]);
        }
        __syncthreads();
    }

    // ---- fused Gabor epilogue ----
    // lin  = Y1 + b1[idx[c]];  lin2 = Y2 + b2[idx[c]]
    // out  = exp(-30*li - 25*(|lin|^2+|lin2|^2)) * (cos(30*lr) + i sin(30*lr))
    const int ci = idx[c];
#pragma unroll
    for (int nt = 0; nt < 2; ++nt) {
        const int f = nBase + nt * 16 + lr16;
        float2 bb1 = b1[ci * OUT_F + f];
        float2 bb2 = b2[ci * OUT_F + f];
#pragma unroll
        for (int v = 0; v < 8; ++v) {
            int row   = mBase + wave * 16 + half * 8 + v;  // C/D layout
            float lr_ = acc[nt][0][v] + bb1.x;
            float li_ = acc[nt][1][v] + bb1.y;
            float mr_ = acc[nt][2][v] + bb2.x;
            float mi_ = acc[nt][3][v] + bb2.y;
            float mag = lr_ * lr_ + li_ * li_ + mr_ * mr_ + mi_ * mi_;
            float are = -30.0f * li_ - 25.0f * mag;
            float aim = 30.0f * lr_;
            float e = __expf(are);
            float s, cn;
            __sincosf(aim, &s, &cn);
            out[((size_t)c * NPTS + row) * OUT_F + f] = make_float2(e * cn, e * s);
        }
    }
}

// ---------------------------------------------------------------------------
extern "C" void kernel_launch(void* const* d_in, const int* in_sizes, int n_in,
                              void* d_out, int out_size, void* d_ws, size_t ws_size,
                              hipStream_t stream)
{
    (void)in_sizes; (void)n_in; (void)out_size; (void)ws_size;

    const float2* x   = (const float2*)d_in[0];
    const int*    idx = (const int*)  d_in[1];
    const float2* U1  = (const float2*)d_in[2];
    const float2* V1  = (const float2*)d_in[3];
    const float2* b1  = (const float2*)d_in[4];
    const float2* U2  = (const float2*)d_in[5];
    const float2* V2  = (const float2*)d_in[6];
    const float2* b2  = (const float2*)d_in[7];
    float2* out = (float2*)d_out;

    // Workspace: 4 bf16 planes of [64][256][256] (transposed weights) = 32 MiB
    bf16_t* Wall = (bf16_t*)d_ws;

    gabor_build_weights<<<dim3(NCHAN * 16), dim3(256), 0, stream>>>(
        idx, U1, V1, U2, V2, Wall);

    gabor_main<<<dim3(NCHAN * 16, OUT_F / 32), dim3(256), 0, stream>>>(
        x, idx, b1, b2, Wall, out);
}
